// TransformerBlock_43026982371492
// MI455X (gfx1250) — matrix operand order repaired and checked by reading, not compiled, not run on hardware
//
#include <hip/hip_runtime.h>

#define DIM     768
#define NHEADS  12
#define HD      64
#define SEQ     2048
#define BATCH   2
#define NEXP    8
#define HID     3072
#define NTOK    (BATCH * SEQ)

// ---------------------------------------------------------------- bf16 utils
__device__ __forceinline__ unsigned short f2bf(float f) {
  union { float f; unsigned u; } v; v.f = f;
  unsigned r = v.u + 0x7FFFu + ((v.u >> 16) & 1u);
  return (unsigned short)(r >> 16);
}
__device__ __forceinline__ float bf2f(unsigned short h) {
  union { unsigned u; float f; } v; v.u = ((unsigned)h) << 16;
  return v.f;
}

typedef __attribute__((ext_vector_type(16))) __bf16 v16bf;
typedef __attribute__((ext_vector_type(8)))  float  v8f;
typedef __attribute__((ext_vector_type(4)))  unsigned v4u;
typedef __attribute__((ext_vector_type(8)))  int      v8i;
union Frag { v16bf v; unsigned u[8]; };

// 16-bit A-matrix 16x32 K-pair offset for dword r, lane half (contiguous k = 16*hlf + 2*r, matches B)
__device__ __forceinline__ int apat(int r, int hlf) {
  return 16 * hlf + 2 * r;
}

__device__ __forceinline__ unsigned ldsoff(const void* p) {
  return (unsigned)(unsigned long long)p;  // LDS aperture: addr[31:0] = offset
}

// ---------------------------------------------------------------- TDM 2-D tile load
// Loads a [tile_h x tile_w] bf16 tile (row stride = stride_elems) from global
// into LDS with 1 pad dword inserted every 2^(pad_code+1) dwords.
// D# packing per CDNA5 ISA 8.3/8.4 (2-D: groups 2/3 omitted -> 2-operand form).
__device__ __forceinline__ void tdm_load_2d_bf16(unsigned lds_addr,
                                                 const void* gptr,
                                                 unsigned tile_w, unsigned tile_h,
                                                 unsigned stride_elems,
                                                 unsigned pad_code) {
  unsigned long long ga = (unsigned long long)gptr;
  v4u g0;
  g0[0] = 1u;                                       // count=1 (valid user D#)
  g0[1] = lds_addr;                                 // lds_addr [63:32]
  g0[2] = (unsigned)(ga & 0xFFFFFFFFu);             // global_addr lo
  g0[3] = (unsigned)((ga >> 32) & 0x01FFFFFFu) | (2u << 30);  // ga hi | type=2
  unsigned td0 = stride_elems;                      // tensor_dim0 >= tile_w
  unsigned td1 = 0x100000u;                         // tensor_dim1: large
  v8i g1;
  g1[0] = (int)((1u << 16) | (1u << 20) | (pad_code << 22));  // 2B elems, pad_en
  g1[1] = (int)((td0 & 0xFFFFu) << 16);                       // tensor_dim0 lo16
  g1[2] = (int)(((td0 >> 16) & 0xFFFFu) | ((td1 & 0xFFFFu) << 16));
  g1[3] = (int)(((td1 >> 16) & 0xFFFFu) | ((tile_w & 0xFFFFu) << 16));
  g1[4] = (int)(tile_h & 0xFFFFu);                  // tile_dim1 (tile_dim2=0)
  g1[5] = (int)stride_elems;                        // tensor_dim0_stride lo32
  g1[6] = 0;
  g1[7] = 0;
  asm volatile("tensor_load_to_lds %0, %1" ::"s"(g0), "s"(g1) : "memory");
}

// ---------------------------------------------------------------- GEMM
// C[M,N] = A[M,K](bf16) x B[K,N](fp32->bf16), fp32 accum. Block: 8 waves as
// 2(M)x4(N); wave tile 32x32 (4 WMMA / K-step); block tile 64x128.
// A tiles arrive via TDM (double buffered); B tiles converted while staging.
// MODE 0: outF = acc          MODE 1: outB = bf16(acc)
// MODE 2: outF = acc + res    MODE 3: h=relu(aux); outB = bf16(h*h*acc)
// MODE 4: outF += rowscale[row*8]*acc     (rowscale also gates a tile skip)
#define GTM 64
#define GTK 32
#define GBN 128
#define LDAS 34
#define LDBS 34

template <int MODE>
__global__ __launch_bounds__(256) void gemm_bf16_wmma(
    const unsigned short* __restrict__ A, const float* __restrict__ B,
    float* __restrict__ outF, unsigned short* __restrict__ outB,
    const unsigned short* __restrict__ aux, const float* __restrict__ res,
    const float* __restrict__ rowscale, int K, int lda, int ldb, int ldc) {
  __shared__ unsigned short As[2][GTM * LDAS];
  __shared__ unsigned short Bt[2][GBN * LDBS];  // transposed: [n][k]
  __shared__ int anyNZ;

  const int tid  = threadIdx.x;
  const int lane = tid & 31;
  const int wv   = tid >> 5;
  const int wm   = wv & 1;
  const int wn   = wv >> 1;
  const int hlf  = (lane >> 4) & 1;
  const int l15  = lane & 15;
  const int row0 = blockIdx.y * GTM;
  const int col0 = blockIdx.x * GBN;

  if (rowscale) {  // skip tile if no row in it routed to this expert
    if (tid == 0) anyNZ = 0;
    __syncthreads();
    if (tid < GTM && rowscale[(size_t)(row0 + tid) * NEXP] != 0.f) anyNZ = 1;
    __syncthreads();
    if (!anyNZ) return;
  }

  // ---- prologue: prefetch k0 = 0 into buffer 0
  if (wv == 0)
    tdm_load_2d_bf16(ldsoff(&As[0][0]), A + (size_t)row0 * lda, GTK, GTM, lda, 3);
  for (int i = tid; i < GTK * GBN; i += 256) {
    int kk = i >> 7, nn = i & 127;
    Bt[0][nn * LDBS + kk] = f2bf(B[(size_t)kk * ldb + col0 + nn]);
  }
  __builtin_amdgcn_s_wait_tensorcnt(0);
  __syncthreads();

  v8f acc[2][2] = {{{}, {}}, {{}, {}}};
  int cur = 0;
  for (int k0 = 0; k0 < K; k0 += GTK) {
    const int nxt = cur ^ 1;
    if (k0 + GTK < K) {  // prefetch next K-tile into the other buffer
      if (wv == 0)
        tdm_load_2d_bf16(ldsoff(&As[nxt][0]),
                         A + (size_t)row0 * lda + (k0 + GTK), GTK, GTM, lda, 3);
      for (int i = tid; i < GTK * GBN; i += 256) {
        int kk = i >> 7, nn = i & 127;
        Bt[nxt][nn * LDBS + kk] =
            f2bf(B[(size_t)(k0 + GTK + kk) * ldb + col0 + nn]);
      }
    }
    const unsigned short* Ac = &As[cur][0];
    const unsigned short* Bc = &Bt[cur][0];
    Frag a0, a1, b0, b1;
    const unsigned short* ar0 = &Ac[(wm * 32 + l15) * LDAS];
    const unsigned short* ar1 = &Ac[(wm * 32 + 16 + l15) * LDAS];
    const unsigned short* br0 = &Bc[(wn * 32 + l15) * LDBS + hlf * 16];
    const unsigned short* br1 = &Bc[(wn * 32 + 16 + l15) * LDBS + hlf * 16];
#pragma unroll
    for (int r = 0; r < 8; ++r) {
      a0.u[r] = *(const unsigned*)(ar0 + apat(r, hlf));
      a1.u[r] = *(const unsigned*)(ar1 + apat(r, hlf));
      b0.u[r] = *(const unsigned*)(br0 + 2 * r);
      b1.u[r] = *(const unsigned*)(br1 + 2 * r);
    }
    acc[0][0] = __builtin_amdgcn_wmma_f32_16x16x32_bf16(false, a0.v, false, b0.v,
                                                        (short)0, acc[0][0], false, false);
    acc[0][1] = __builtin_amdgcn_wmma_f32_16x16x32_bf16(false, a0.v, false, b1.v,
                                                        (short)0, acc[0][1], false, false);
    acc[1][0] = __builtin_amdgcn_wmma_f32_16x16x32_bf16(false, a1.v, false, b0.v,
                                                        (short)0, acc[1][0], false, false);
    acc[1][1] = __builtin_amdgcn_wmma_f32_16x16x32_bf16(false, a1.v, false, b1.v,
                                                        (short)0, acc[1][1], false, false);
    __builtin_amdgcn_s_wait_tensorcnt(0);
    __syncthreads();
    cur = nxt;
  }

#pragma unroll
  for (int i = 0; i < 2; ++i)
#pragma unroll
    for (int j = 0; j < 2; ++j) {
      const int rbase = row0 + wm * 32 + i * 16 + hlf * 8;
      const int cn    = col0 + wn * 32 + j * 16 + l15;
#pragma unroll
      for (int r = 0; r < 8; ++r) {
        size_t idx = (size_t)(rbase + r) * ldc + cn;
        float v = acc[i][j][r];
        if (MODE == 0) outF[idx] = v;
        if (MODE == 1) outB[idx] = f2bf(v);
        if (MODE == 2) outF[idx] = v + res[idx];
        if (MODE == 3) {
          float h = bf2f(aux[idx]); h = h > 0.f ? h : 0.f;
          outB[idx] = f2bf(h * h * v);
        }
        if (MODE == 4) {
          float s = rowscale[(size_t)(rbase + r) * NEXP];
          if (s != 0.f) outF[idx] += s * v;
        }
      }
    }
}

// ---------------------------------------------------------------- RMSNorm
__global__ __launch_bounds__(256) void rmsnorm_kernel(
    const float* __restrict__ x, const float* __restrict__ w,
    float* __restrict__ outF, unsigned short* __restrict__ outB) {
  __shared__ float red[256];
  const int row = blockIdx.x, tid = threadIdx.x;
  const float* xr = x + (size_t)row * DIM;
  float ss = 0.f;
  for (int i = tid; i < DIM; i += 256) { float v = xr[i]; ss += v * v; }
  red[tid] = ss; __syncthreads();
  for (int s = 128; s > 0; s >>= 1) {
    if (tid < s) red[tid] += red[tid + s];
    __syncthreads();
  }
  const float scale = rsqrtf(red[0] / (float)DIM + 1e-6f);
  for (int i = tid; i < DIM; i += 256) {
    float v = xr[i] * scale * w[i];
    if (outF) outF[(size_t)row * DIM + i] = v;
    outB[(size_t)row * DIM + i] = f2bf(v);
  }
}

// ---------------------------------------------------------------- RoPE
__global__ void rope_kernel(const float* __restrict__ Qf,
                            const float* __restrict__ Kf,
                            unsigned short* __restrict__ Qb,
                            unsigned short* __restrict__ Kb) {
  int idx = blockIdx.x * blockDim.x + threadIdx.x;
  if (idx >= NTOK * DIM) return;
  int row = idx / DIM, hd = idx % DIM;
  int s = row % SEQ;
  int d = hd & (HD - 1);
  int dh = d & 31;
  float inv = __powf(10000.f, -(float)(2 * dh) / (float)HD);
  float th = (float)s * inv;
  float c = __cosf(th), sn = __sinf(th);
  float q = Qf[idx], k = Kf[idx];
  float qr = (d < 32) ? -Qf[idx + 32] : Qf[idx - 32];
  float kr = (d < 32) ? -Kf[idx + 32] : Kf[idx - 32];
  Qb[idx] = f2bf(q * c + qr * sn);
  Kb[idx] = f2bf(k * c + kr * sn);
}

// ---------------------------------------------------------------- Flash attention
// Block = 128 threads (4 waves); each wave owns 16 q rows; key blocks of 32.
// K block arrives via TDM (row pad -> 66-ushort stride); V staged transposed.
__global__ __launch_bounds__(128) void flash_attn(
    const unsigned short* __restrict__ Qb, const unsigned short* __restrict__ Kb,
    const unsigned short* __restrict__ Vb, const float* __restrict__ mask,
    unsigned short* __restrict__ ctx) {
  __shared__ unsigned short Kl[32 * 66];       // [key][d]  (TDM, pad_code=4)
  __shared__ unsigned short Vl[64 * 34];       // [d][key]  (transposed)
  __shared__ unsigned short Pl[4 * 16 * 34];   // per-wave P staging
  const int tid  = threadIdx.x;
  const int lane = tid & 31;
  const int wv   = tid >> 5;
  const int hlf  = (lane >> 4) & 1;
  const int l15  = lane & 15;
  const int q0 = blockIdx.x * 64, h = blockIdx.y, b = blockIdx.z;
  const int qrow = q0 + wv * 16;
  const int ldq = NHEADS * HD;

  Frag qf[2];
#pragma unroll
  for (int c = 0; c < 2; ++c)
#pragma unroll
    for (int r = 0; r < 8; ++r)
      qf[c].u[r] = *(const unsigned*)(Qb + (size_t)(b * SEQ + qrow + l15) * ldq +
                                      h * HD + c * 32 + apat(r, hlf));

  float mrun[8], lrun[8];
  v8f o[4] = {{}, {}, {}, {}};
#pragma unroll
  for (int r = 0; r < 8; ++r) { mrun[r] = -1e30f; lrun[r] = 0.f; }
  unsigned short* Pw = &Pl[wv * 16 * 34];

  for (int kb = 0; kb < SEQ; kb += 32) {
    __syncthreads();
    if (wv == 0)
      tdm_load_2d_bf16(ldsoff(&Kl[0]),
                       Kb + (size_t)(b * SEQ + kb) * ldq + h * HD,
                       HD, 32, ldq, 4);
    for (int i = tid; i < 2048; i += 128) {  // V block transposed
      int kr = i >> 6, d = i & 63;
      Vl[d * 34 + kr] = Vb[(size_t)(b * SEQ + kb + kr) * ldq + h * HD + d];
    }
    __builtin_amdgcn_s_wait_tensorcnt(0);
    __syncthreads();

    v8f s0 = {}, s1 = {};
    Frag bf;
#pragma unroll
    for (int c = 0; c < 2; ++c) {
#pragma unroll
      for (int r = 0; r < 8; ++r)
        bf.u[r] = *(const unsigned*)(&Kl[l15 * 66 + c * 32 + hlf * 16 + 2 * r]);
      s0 = __builtin_amdgcn_wmma_f32_16x16x32_bf16(false, qf[c].v, false, bf.v,
                                                   (short)0, s0, false, false);
#pragma unroll
      for (int r = 0; r < 8; ++r)
        bf.u[r] = *(const unsigned*)(&Kl[(16 + l15) * 66 + c * 32 + hlf * 16 + 2 * r]);
      s1 = __builtin_amdgcn_wmma_f32_16x16x32_bf16(false, qf[c].v, false, bf.v,
                                                   (short)0, s1, false, false);
    }

    const int myrow = qrow + hlf * 8;
#pragma unroll
    for (int r = 0; r < 8; ++r) {
      float sa = s0[r] * 0.125f + mask[(size_t)(myrow + r) * SEQ + kb + l15];
      float sb = s1[r] * 0.125f + mask[(size_t)(myrow + r) * SEQ + kb + 16 + l15];
      float mx = fmaxf(sa, sb);
#pragma unroll
      for (int off = 8; off >= 1; off >>= 1) mx = fmaxf(mx, __shfl_xor(mx, off, 16));
      float mnew = fmaxf(mrun[r], mx);
      float corr = __expf(mrun[r] - mnew);
      float p0 = __expf(sa - mnew);
      float p1 = __expf(sb - mnew);
      float rs = p0 + p1;
#pragma unroll
      for (int off = 8; off >= 1; off >>= 1) rs += __shfl_xor(rs, off, 16);
      lrun[r] = lrun[r] * corr + rs;
      mrun[r] = mnew;
#pragma unroll
      for (int t = 0; t < 4; ++t) o[t][r] *= corr;
      Pw[(r + hlf * 8) * 34 + l15]      = f2bf(p0);
      Pw[(r + hlf * 8) * 34 + 16 + l15] = f2bf(p1);
    }
    __syncthreads();

    Frag pf;
#pragma unroll
    for (int r = 0; r < 8; ++r)
      pf.u[r] = *(const unsigned*)(&Pw[l15 * 34 + apat(r, hlf)]);
#pragma unroll
    for (int t = 0; t < 4; ++t) {
      Frag vf;
#pragma unroll
      for (int r = 0; r < 8; ++r)
        vf.u[r] = *(const unsigned*)(&Vl[(16 * t + l15) * 34 + hlf * 16 + 2 * r]);
      o[t] = __builtin_amdgcn_wmma_f32_16x16x32_bf16(false, pf.v, false, vf.v,
                                                     (short)0, o[t], false, false);
    }
  }

#pragma unroll
  for (int r = 0; r < 8; ++r) {
    float inv = 1.f / lrun[r];
    int orow = qrow + hlf * 8 + r;
#pragma unroll
    for (int t = 0; t < 4; ++t)
      ctx[(size_t)(b * SEQ + orow) * ldq + h * HD + 16 * t + l15] =
          f2bf(o[t][r] * inv);
  }
}

// ---------------------------------------------------------------- gate + top-2
__global__ __launch_bounds__(256) void gate_kernel(
    const float* __restrict__ xn, const float* __restrict__ gw,
    const float* __restrict__ gb, float* __restrict__ cw) {
  __shared__ float red[256];
  __shared__ float logits[NEXP];
  const int row = blockIdx.x, tid = threadIdx.x;
  float part[NEXP];
#pragma unroll
  for (int e = 0; e < NEXP; ++e) part[e] = 0.f;
  for (int i = tid; i < DIM; i += 256) {
    float xv = xn[(size_t)row * DIM + i];
#pragma unroll
    for (int e = 0; e < NEXP; ++e) part[e] += xv * gw[i * NEXP + e];
  }
  for (int e = 0; e < NEXP; ++e) {
    red[tid] = part[e]; __syncthreads();
    for (int s = 128; s > 0; s >>= 1) {
      if (tid < s) red[tid] += red[tid + s];
      __syncthreads();
    }
    if (tid == 0) logits[e] = red[0] + gb[e];
    __syncthreads();
  }
  if (tid == 0) {
    float mx = logits[0];
    for (int e = 1; e < NEXP; ++e) mx = fmaxf(mx, logits[e]);
    float p[NEXP];
    for (int e = 0; e < NEXP; ++e) p[e] = __expf(logits[e] - mx);
    int i0 = 0;
    for (int e = 1; e < NEXP; ++e) if (p[e] > p[i0]) i0 = e;
    int i1 = (i0 == 0) ? 1 : 0;
    for (int e = 0; e < NEXP; ++e) if (e != i0 && p[e] > p[i1]) i1 = e;
    float s2 = p[i0] + p[i1];
    for (int e = 0; e < NEXP; ++e) cw[(size_t)row * NEXP + e] = 0.f;
    cw[(size_t)row * NEXP + i0] = p[i0] / s2;
    cw[(size_t)row * NEXP + i1] = p[i1] / s2;
  }
}

// ---------------------------------------------------------------- launch
extern "C" void kernel_launch(void* const* d_in, const int* in_sizes, int n_in,
                              void* d_out, int out_size, void* d_ws, size_t ws_size,
                              hipStream_t stream) {
  const float* x    = (const float*)d_in[0];
  const float* mask = (const float*)d_in[1];
  const float* n1w  = (const float*)d_in[2];
  const float* n2w  = (const float*)d_in[3];
  const float* wq   = (const float*)d_in[4];
  const float* wk   = (const float*)d_in[5];
  const float* wv   = (const float*)d_in[6];
  const float* wo   = (const float*)d_in[7];
  const float* gw   = (const float*)d_in[8];
  const float* gb   = (const float*)d_in[9];
  const float* w1   = (const float*)d_in[10];
  const float* w3   = (const float*)d_in[11];
  const float* w2   = (const float*)d_in[12];
  float* out = (float*)d_out;

  char* p = (char*)d_ws;
  auto alloc = [&](size_t bytes) -> void* {
    void* r = (void*)p;
    p += (bytes + 255) & ~(size_t)255;
    return r;
  };
  unsigned short* xn1b = (unsigned short*)alloc((size_t)NTOK * DIM * 2);
  float*          Qf   = (float*)         alloc((size_t)NTOK * DIM * 4);
  float*          Kf   = (float*)         alloc((size_t)NTOK * DIM * 4);
  unsigned short* Qb   = (unsigned short*)alloc((size_t)NTOK * DIM * 2);
  unsigned short* Kb   = (unsigned short*)alloc((size_t)NTOK * DIM * 2);
  unsigned short* Vb   = (unsigned short*)alloc((size_t)NTOK * DIM * 2);
  unsigned short* ctxb = (unsigned short*)alloc((size_t)NTOK * DIM * 2);
  float*          xn2f = (float*)         alloc((size_t)NTOK * DIM * 4);
  unsigned short* xn2b = (unsigned short*)alloc((size_t)NTOK * DIM * 2);
  float*          cw   = (float*)         alloc((size_t)NTOK * NEXP * 4);
  unsigned short* h1b  = (unsigned short*)alloc((size_t)NTOK * HID * 2);
  unsigned short* gbf  = (unsigned short*)alloc((size_t)NTOK * HID * 2);

  const dim3 blk(256);
  const dim3 gq(DIM / GBN, NTOK / GTM);   // N=768 GEMMs (6 x 64)
  const dim3 gh(HID / GBN, NTOK / GTM);   // N=3072 GEMMs (24 x 64)

  // 1) norm1 -> bf16
  rmsnorm_kernel<<<NTOK, blk, 0, stream>>>(x, n1w, nullptr, xn1b);
  // 2) Q,K fp32 (for RoPE); V straight to bf16
  gemm_bf16_wmma<0><<<gq, blk, 0, stream>>>(xn1b, wq, Qf, nullptr, nullptr, nullptr, nullptr, DIM, DIM, DIM, DIM);
  gemm_bf16_wmma<0><<<gq, blk, 0, stream>>>(xn1b, wk, Kf, nullptr, nullptr, nullptr, nullptr, DIM, DIM, DIM, DIM);
  gemm_bf16_wmma<1><<<gq, blk, 0, stream>>>(xn1b, wv, nullptr, Vb, nullptr, nullptr, nullptr, DIM, DIM, DIM, DIM);
  // 3) RoPE -> bf16 Q,K
  rope_kernel<<<(NTOK * DIM + 255) / 256, blk, 0, stream>>>(Qf, Kf, Qb, Kb);
  // 4) flash attention -> bf16 ctx
  flash_attn<<<dim3(SEQ / 64, NHEADS, BATCH), dim3(128), 0, stream>>>(Qb, Kb, Vb, mask, ctxb);
  // 5) output projection + residual -> d_out (= x1)
  gemm_bf16_wmma<2><<<gq, blk, 0, stream>>>(ctxb, wo, out, nullptr, nullptr, x, nullptr, DIM, DIM, DIM, DIM);
  // 6) norm2 (reads x1 from d_out)
  rmsnorm_kernel<<<NTOK, blk, 0, stream>>>(out, n2w, xn2f, xn2b);
  // 7) gate -> dense combine weights cw[N, 8]
  gate_kernel<<<NTOK, blk, 0, stream>>>(xn2f, gw, gb, cw);
  // 8) experts: h1 = xn2@w1 (bf16); g = relu(h1)^2 * (xn2@w3) (bf16); out += cw*(g@w2)
  for (int e = 0; e < NEXP; ++e) {
    const float* w1e = w1 + (size_t)e * DIM * HID;
    const float* w3e = w3 + (size_t)e * DIM * HID;
    const float* w2e = w2 + (size_t)e * HID * DIM;
    const float* rs  = cw + e;
    gemm_bf16_wmma<1><<<gh, blk, 0, stream>>>(xn2b, w1e, nullptr, h1b, nullptr, nullptr, rs, DIM, DIM, HID, HID);
    gemm_bf16_wmma<3><<<gh, blk, 0, stream>>>(xn2b, w3e, nullptr, gbf, h1b, nullptr, rs, DIM, DIM, HID, HID);
    gemm_bf16_wmma<4><<<gq, blk, 0, stream>>>(gbf, w2e, out, nullptr, nullptr, nullptr, rs, HID, HID, DIM, DIM);
  }
}
